// SparseFlowMLP_82497731822061
// MI455X (gfx1250) — matrix operand order repaired and checked by reading, not compiled, not run on hardware
//
#include <hip/hip_runtime.h>
#include <hip/hip_bf16.h>

// MI455X fused 2:4-pruned SwiGLU MLP.
// Stage 1: hidden = silu(x@prune24(gw).T + gb) * (x@prune24(uw).T + ub)  (bf16, token-major, in d_ws)
// Stage 2: out    = hidden @ dw.T + db                                   (fp32)
// Memory-bound: 541 MB of fp32 weights streamed exactly once; WMMA bf16 for all matmuls.

typedef __bf16 bf16_t;
typedef bf16_t v16bf __attribute__((ext_vector_type(16)));
typedef float  v8f   __attribute__((ext_vector_type(8)));

#define H_DIM 4096
#define I_DIM 11008

static __device__ __forceinline__ float4 ldg4(const float* p) {
  return *reinterpret_cast<const float4*>(p);
}

// 2:4 magnitude pruning of one contiguous group of 4 along K.
// Matches jnp.argsort (stable ascending): keeps the 2 largest |w|,
// ties broken toward the higher index.
static __device__ __forceinline__ float4 prune4(float4 v) {
  float w[4] = {v.x, v.y, v.z, v.w};
  float a[4] = {__builtin_fabsf(v.x), __builtin_fabsf(v.y),
                __builtin_fabsf(v.z), __builtin_fabsf(v.w)};
#pragma unroll
  for (int i = 0; i < 4; ++i) {
    int beat = 0;
#pragma unroll
    for (int j = 0; j < 4; ++j) {
      if (j == i) continue;
      beat += (a[j] > a[i]) || ((a[j] == a[i]) && (j > i));
    }
    if (beat >= 2) w[i] = 0.0f;
  }
  return make_float4(w[0], w[1], w[2], w[3]);
}

// Pack 16 floats (K-ascending per the A/B fragment layouts) into a bf16 fragment.
static __device__ __forceinline__ v16bf pack16(float4 a, float4 b, float4 c, float4 d) {
  v16bf r;
  r[ 0]=(bf16_t)a.x; r[ 1]=(bf16_t)a.y; r[ 2]=(bf16_t)a.z; r[ 3]=(bf16_t)a.w;
  r[ 4]=(bf16_t)b.x; r[ 5]=(bf16_t)b.y; r[ 6]=(bf16_t)b.z; r[ 7]=(bf16_t)b.w;
  r[ 8]=(bf16_t)c.x; r[ 9]=(bf16_t)c.y; r[10]=(bf16_t)c.z; r[11]=(bf16_t)c.w;
  r[12]=(bf16_t)d.x; r[13]=(bf16_t)d.y; r[14]=(bf16_t)d.z; r[15]=(bf16_t)d.w;
  return r;
}

static __device__ __forceinline__ v8f wmma_bf16(v16bf a, v16bf b, v8f c) {
  // (neg_a, A, neg_b, B, c_mod, C, reuse_a, reuse_b)
  return __builtin_amdgcn_wmma_f32_16x16x32_bf16(false, a, false, b, (short)0, c,
                                                 false, false);
}

// ---------------------------------------------------------------------------
// Stage 1: gate/up projections with on-the-fly 2:4 pruning + SiLU fusion.
// Wave tile: 16 I-rows x 32 tokens (two N-tiles share the A fragment).
// Block = 8 waves: 4 I-tiles x K-split of 2, LDS reduction.
// ---------------------------------------------------------------------------
__global__ __launch_bounds__(256)
void gate_up_silu_kernel(const float* __restrict__ x,
                         const float* __restrict__ gw,
                         const float* __restrict__ uw,
                         const float* __restrict__ gb,
                         const float* __restrict__ ub,
                         bf16_t* __restrict__ hidden) {
  __shared__ float red[4][4][32][8];  // [i-tile pair][acc][lane][row] = 16 KB

  const int lane  = threadIdx.x & 31;
  const int wave  = threadIdx.x >> 5;
  const int pair  = wave >> 1;                      // which of 4 I-tiles in block
  const int kq    = wave & 1;                       // K-split index
  const int i0    = (blockIdx.x * 4 + pair) * 16;   // grid 172 -> 688 tiles
  const int mrow  = lane & 15;
  const int khalf = lane >> 4;

  const float* grow  = gw + (size_t)(i0 + mrow) * H_DIM;
  const float* urow  = uw + (size_t)(i0 + mrow) * H_DIM;
  const float* xrow0 = x  + (size_t)(mrow)      * H_DIM;  // tokens 0..15
  const float* xrow1 = x  + (size_t)(mrow + 16) * H_DIM;  // tokens 16..31

  v8f cg0 = {}, cg1 = {}, cu0 = {}, cu1 = {};

  const int kbeg = kq * (H_DIM / 2);
  const int kend = kbeg + (H_DIM / 2);
  for (int k0 = kbeg; k0 < kend; k0 += 32) {
    __builtin_prefetch(grow + k0 + 1024, 0, 0);   // global_prefetch_b8
    __builtin_prefetch(urow + k0 + 1024, 0, 0);
    const int ka = k0 + khalf * 8;                // A frag: K quads per ISA layout
    const int kb = ka + 16;
    v16bf ag = pack16(prune4(ldg4(grow + ka)), prune4(ldg4(grow + ka + 4)),
                      prune4(ldg4(grow + kb)), prune4(ldg4(grow + kb + 4)));
    v16bf au = pack16(prune4(ldg4(urow + ka)), prune4(ldg4(urow + ka + 4)),
                      prune4(ldg4(urow + kb)), prune4(ldg4(urow + kb + 4)));
    // B frag: same K quads as A
    v16bf b0 = pack16(ldg4(xrow0 + ka),     ldg4(xrow0 + ka + 4),
                      ldg4(xrow0 + kb), ldg4(xrow0 + kb + 4));
    v16bf b1 = pack16(ldg4(xrow1 + ka),     ldg4(xrow1 + ka + 4),
                      ldg4(xrow1 + kb), ldg4(xrow1 + kb + 4));
    cg0 = wmma_bf16(ag, b0, cg0);
    cg1 = wmma_bf16(ag, b1, cg1);
    cu0 = wmma_bf16(au, b0, cu0);
    cu1 = wmma_bf16(au, b1, cu1);
  }

  // K-split reduction across the wave pair.
  if (kq == 1) {
#pragma unroll
    for (int r = 0; r < 8; ++r) {
      red[pair][0][lane][r] = cg0[r];
      red[pair][1][lane][r] = cg1[r];
      red[pair][2][lane][r] = cu0[r];
      red[pair][3][lane][r] = cu1[r];
    }
  }
  __syncthreads();
  if (kq == 1) return;
#pragma unroll
  for (int r = 0; r < 8; ++r) {
    cg0[r] += red[pair][0][lane][r];
    cg1[r] += red[pair][1][lane][r];
    cu0[r] += red[pair][2][lane][r];
    cu1[r] += red[pair][3][lane][r];
  }

  // Epilogue: bias + SiLU*up, store bf16 hidden[token][i] (two 16B stores/lane).
  const int ibase = i0 + 8 * khalf;               // C/D: elem r -> M = r + 8*khalf
  float4 gl = ldg4(gb + ibase), gh = ldg4(gb + ibase + 4);
  float4 ul = ldg4(ub + ibase), uh = ldg4(ub + ibase + 4);
  float gbv[8] = {gl.x, gl.y, gl.z, gl.w, gh.x, gh.y, gh.z, gh.w};
  float ubv[8] = {ul.x, ul.y, ul.z, ul.w, uh.x, uh.y, uh.z, uh.w};

  union { bf16_t h[8]; uint4 q; } o0, o1;
#pragma unroll
  for (int r = 0; r < 8; ++r) {
    float g0 = cg0[r] + gbv[r];
    float u0 = cu0[r] + ubv[r];
    o0.h[r] = (bf16_t)((g0 / (1.0f + __expf(-g0))) * u0);
    float g1 = cg1[r] + gbv[r];
    float u1 = cu1[r] + ubv[r];
    o1.h[r] = (bf16_t)((g1 / (1.0f + __expf(-g1))) * u1);
  }
  *reinterpret_cast<uint4*>(hidden + (size_t)(mrow)      * I_DIM + ibase) = o0.q;
  *reinterpret_cast<uint4*>(hidden + (size_t)(mrow + 16) * I_DIM + ibase) = o1.q;
}

// ---------------------------------------------------------------------------
// Stage 2: out = hidden @ dw.T + db.  Wave tile: 16 H-rows x 32 tokens.
// Block = 8 waves: 2 H-tiles x K-split of 4 (11008/4 = 2752), LDS reduction.
// ---------------------------------------------------------------------------
__global__ __launch_bounds__(256)
void down_proj_kernel(const float* __restrict__ dw,
                      const bf16_t* __restrict__ hidden,
                      const float* __restrict__ db,
                      float* __restrict__ out) {
  __shared__ float red[2][3][2][32][8];  // [grp][kq-1][acc][lane][row] = 12 KB

  const int lane  = threadIdx.x & 31;
  const int wave  = threadIdx.x >> 5;
  const int grp   = wave >> 2;
  const int kq    = wave & 3;
  const int h0    = (blockIdx.x * 2 + grp) * 16;   // grid 128 -> 256 tiles
  const int mrow  = lane & 15;
  const int khalf = lane >> 4;

  const float*  drow = dw     + (size_t)(h0 + mrow) * I_DIM;
  const bf16_t* hr0  = hidden + (size_t)(mrow)      * I_DIM;
  const bf16_t* hr1  = hidden + (size_t)(mrow + 16) * I_DIM;

  v8f c0 = {}, c1 = {};
  const int kbeg = kq * (I_DIM / 4);
  const int kend = kbeg + (I_DIM / 4);
  for (int k0 = kbeg; k0 < kend; k0 += 32) {
    __builtin_prefetch(drow + k0 + 1024, 0, 0);
    const int ka = k0 + khalf * 8;
    const int kb = ka + 16;
    v16bf a = pack16(ldg4(drow + ka), ldg4(drow + ka + 4),
                     ldg4(drow + kb), ldg4(drow + kb + 4));
    const int kx = k0 + khalf * 16;
    union { uint4 q[2]; v16bf v; } b0u, b1u;
    b0u.q[0] = *reinterpret_cast<const uint4*>(hr0 + ka);
    b0u.q[1] = *reinterpret_cast<const uint4*>(hr0 + kb);
    b1u.q[0] = *reinterpret_cast<const uint4*>(hr1 + ka);
    b1u.q[1] = *reinterpret_cast<const uint4*>(hr1 + kb);
    c0 = wmma_bf16(a, b0u.v, c0);
    c1 = wmma_bf16(a, b1u.v, c1);
  }

  if (kq != 0) {
#pragma unroll
    for (int r = 0; r < 8; ++r) {
      red[grp][kq - 1][0][lane][r] = c0[r];
      red[grp][kq - 1][1][lane][r] = c1[r];
    }
  }
  __syncthreads();
  if (kq != 0) return;
#pragma unroll
  for (int q = 0; q < 3; ++q)
#pragma unroll
    for (int r = 0; r < 8; ++r) {
      c0[r] += red[grp][q][0][lane][r];
      c1[r] += red[grp][q][1][lane][r];
    }

  const int hbase = h0 + 8 * khalf;
  float4 bl = ldg4(db + hbase), bh = ldg4(db + hbase + 4);
  float bv[8] = {bl.x, bl.y, bl.z, bl.w, bh.x, bh.y, bh.z, bh.w};
  float o0[8], o1[8];
#pragma unroll
  for (int r = 0; r < 8; ++r) {
    o0[r] = c0[r] + bv[r];
    o1[r] = c1[r] + bv[r];
  }
  float* p0 = out + (size_t)(mrow)      * H_DIM + hbase;
  float* p1 = out + (size_t)(mrow + 16) * H_DIM + hbase;
  *reinterpret_cast<float4*>(p0)     = make_float4(o0[0], o0[1], o0[2], o0[3]);
  *reinterpret_cast<float4*>(p0 + 4) = make_float4(o0[4], o0[5], o0[6], o0[7]);
  *reinterpret_cast<float4*>(p1)     = make_float4(o1[0], o1[1], o1[2], o1[3]);
  *reinterpret_cast<float4*>(p1 + 4) = make_float4(o1[4], o1[5], o1[6], o1[7]);
}

extern "C" void kernel_launch(void* const* d_in, const int* in_sizes, int n_in,
                              void* d_out, int out_size, void* d_ws, size_t ws_size,
                              hipStream_t stream) {
  const float* x  = (const float*)d_in[0];   // (8,4,4096)
  const float* gw = (const float*)d_in[1];   // (11008,4096)
  const float* uw = (const float*)d_in[2];   // (11008,4096)
  const float* dw = (const float*)d_in[3];   // (4096,11008)
  const float* gb = (const float*)d_in[4];   // (11008,)
  const float* ub = (const float*)d_in[5];   // (11008,)
  const float* db = (const float*)d_in[6];   // (4096,)
  float*  out    = (float*)d_out;            // (8,4,4096) fp32
  bf16_t* hidden = (bf16_t*)d_ws;            // 32 x 11008 bf16 = 688 KB scratch

  // Stage 1: 688 I-tiles / 4 per block -> 172 blocks of 8 waves.
  gate_up_silu_kernel<<<172, 256, 0, stream>>>(x, gw, uw, gb, ub, hidden);
  // Stage 2: 256 H-tiles / 2 per block -> 128 blocks of 8 waves.
  down_proj_kernel<<<128, 256, 0, stream>>>(dw, hidden, db, out);
}
